// MultiheadAttention_30348238914174
// MI455X (gfx1250) — compile-verified, matrix operand order checked by reading, not run on hardware
//
#include <hip/hip_runtime.h>
#include <math.h>

// ---------------------------------------------------------------------------
// MHA forward for MI455X (gfx1250).  bf16 WMMA for all four GEMMs, flash-style
// 2-pass softmax (256MB score tensor never materialized; ~32MB ws is L2-res).
// Pass 2 groups the 16 (head, n-offset) combos feeding one avg_w tile into a
// 16-wave workgroup -> avg_w reduced in LDS, no global atomics.
// This round: unroll-by-2 ping-pong fragment buffers (prefetch writes directly
// into the retired buffer set) -> no v_mov rotation copies, partial loadcnt
// waits, WMMA overlapped with VMEM.
// ---------------------------------------------------------------------------

typedef __bf16 bf16_t;
typedef __attribute__((ext_vector_type(16))) __bf16 v16bf;
typedef __attribute__((ext_vector_type(8)))  __bf16 v8bf;
typedef __attribute__((ext_vector_type(8)))  float  v8f;

#define E_DIM   1024
#define H_DIM   16
#define D_DIM   64
#define T_DIM   1024
#define B_DIM   2
#define N_TOK   2048            // T*B flattened attention axis
#define SCALING 0.125f          // D^-0.5

__device__ __forceinline__ v8f wmma_bf16(v16bf a, v16bf b, v8f c) {
  // (neg_a, A, neg_b, B, c_mod, C, reuse_a, reuse_b)
  return __builtin_amdgcn_wmma_f32_16x16x32_bf16(false, a, false, b, (short)0, c,
                                                 false, false);
}

// A/B fragment load from K-contiguous bf16 memory (NT GEMM layout).
// CDNA5 16-bit A layout: lane half hh holds K chunks [8hh,8hh+8) and
// [16+8hh,16+8hh+8); row = lane&15.  B per-lane column load from the
// transposed operand's row uses identical addressing.
__device__ __forceinline__ v16bf load_frag_bf16(const bf16_t* __restrict__ base,
                                                int stride, int row0, int col0) {
  int lane = threadIdx.x & 31;
  int r = lane & 15;
  int hh = lane >> 4;
  const bf16_t* p = base + (size_t)(row0 + r) * stride + col0 + 8 * hh;
  v8bf lo = *(const v8bf*)(p);
  v8bf hi = *(const v8bf*)(p + 16);
  v16bf out;
#pragma unroll
  for (int i = 0; i < 8; ++i) { out[i] = lo[i]; out[8 + i] = hi[i]; }
  return out;
}

// A fragment from an fp32 16x32 tile staged in LDS (stride floats), cvt->bf16.
__device__ __forceinline__ v16bf load_frag_lds_f32(const float* __restrict__ p,
                                                   int stride) {
  int lane = threadIdx.x & 31;
  int r = lane & 15;
  int hh = lane >> 4;
  v16bf out;
#pragma unroll
  for (int i = 0; i < 8; ++i) {
    out[i]     = (bf16_t)p[r * stride + 8 * hh + i];
    out[8 + i] = (bf16_t)p[r * stride + 16 + 8 * hh + i];
  }
  return out;
}

// Online softmax update for one 16x16 score tile held in C-layout:
// row m = v + 8*(lane>>4) lives in the 16 lanes of that half-wave.
__device__ __forceinline__ void online_update(const v8f& c, float* rmax,
                                              float* rsum) {
#pragma unroll
  for (int v = 0; v < 8; ++v) {
    float sc = c[v];
    float tmax = sc;                   // reduce over 16 cols (half-wave)
#pragma unroll
    for (int m = 1; m < 16; m <<= 1) tmax = fmaxf(tmax, __shfl_xor(tmax, m, 32));
    float nmax = fmaxf(rmax[v], tmax);
    float e = __expf(sc - nmax);
    float tsum = e;
#pragma unroll
    for (int m = 1; m < 16; m <<= 1) tsum += __shfl_xor(tsum, m, 32);
    rsum[v] = rsum[v] * __expf(rmax[v] - nmax) + tsum;
    rmax[v] = nmax;
  }
}

// ------------------------- elementwise prep kernel -------------------------
__global__ void cvt_f32_to_bf16_kernel(const float* __restrict__ src,
                                       bf16_t* __restrict__ dst, int n) {
  int i = (blockIdx.x * blockDim.x + threadIdx.x) * 4;
  if (i < n) {
#pragma unroll
    for (int j = 0; j < 4; ++j) dst[i + j] = (bf16_t)src[i + j];
  }
}

// ------------------------------ QKV projection -----------------------------
// Proj[n, j] = sum_e X[n,e] * w_in[j,e] + b_in[j];  X = query for j<E else key
// (reference derives BOTH k and v from `key`).  q rows pre-scaled by D^-0.5.
// V stored transposed Vt[h][d][n] so the attn*V B operand is s-contiguous.
// 16x64 per-wave register block; ping-pong double-buffered fragments.
__global__ __launch_bounds__(128) void proj_qkv_kernel(
    const bf16_t* __restrict__ Xq, const bf16_t* __restrict__ Xk,
    const bf16_t* __restrict__ Wbf, const float* __restrict__ b_in,
    bf16_t* __restrict__ Qp, bf16_t* __restrict__ Kp, bf16_t* __restrict__ Vt) {
  int wave = threadIdx.x >> 5;
  int lane = threadIdx.x & 31;
  int j0 = (blockIdx.x * 4 + wave) * 64;   // 48 j-blocks of 64 over 3E
  int n0 = blockIdx.y * 16;
  const bf16_t* X = (j0 < E_DIM) ? Xq : Xk;   // blocks never straddle q|k|v

  v8f acc[4] = {};
  v16bf a0 = load_frag_bf16(X, E_DIM, n0, 0);
  v16bf b0[4];
#pragma unroll
  for (int jt = 0; jt < 4; ++jt) b0[jt] = load_frag_bf16(Wbf, E_DIM, j0 + jt * 16, 0);

  for (int k0 = 0; k0 < E_DIM; k0 += 64) {
    v16bf a1 = load_frag_bf16(X, E_DIM, n0, k0 + 32);
    v16bf b1[4];
#pragma unroll
    for (int jt = 0; jt < 4; ++jt)
      b1[jt] = load_frag_bf16(Wbf, E_DIM, j0 + jt * 16, k0 + 32);
#pragma unroll
    for (int jt = 0; jt < 4; ++jt) acc[jt] = wmma_bf16(a0, b0[jt], acc[jt]);

    int kn = (k0 + 64) & (E_DIM - 1);      // last iter wraps to 0 (discarded)
    a0 = load_frag_bf16(X, E_DIM, n0, kn);
#pragma unroll
    for (int jt = 0; jt < 4; ++jt)
      b0[jt] = load_frag_bf16(Wbf, E_DIM, j0 + jt * 16, kn);
#pragma unroll
    for (int jt = 0; jt < 4; ++jt) acc[jt] = wmma_bf16(a1, b1[jt], acc[jt]);
  }

  int col = lane & 15;
  int rh  = lane >> 4;
#pragma unroll
  for (int jt = 0; jt < 4; ++jt) {
    int j = j0 + jt * 16 + col;
    float bias = b_in[j];
#pragma unroll
    for (int v = 0; v < 8; ++v) {
      int n = n0 + v + 8 * rh;
      float val = acc[jt][v] + bias;
      if (j < E_DIM) {
        Qp[(size_t)n * E_DIM + j] = (bf16_t)(val * SCALING);
      } else if (j < 2 * E_DIM) {
        Kp[(size_t)n * E_DIM + (j - E_DIM)] = (bf16_t)val;
      } else {
        int jj = j - 2 * E_DIM;
        int h = jj >> 6, d = jj & 63;
        Vt[((size_t)(h * D_DIM + d)) * N_TOK + n] = (bf16_t)val;
      }
    }
  }
}

// ----------------------- pass 1: softmax row statistics --------------------
// One wave per (head, 16-row n-tile): online max / sumexp over all 2048 cols.
// Ping-pong K-tile fragments; shuffle/exp VALU overlaps the in-flight loads.
__global__ __launch_bounds__(128) void attn_stats_kernel(
    const bf16_t* __restrict__ Qp, const bf16_t* __restrict__ Kp,
    float* __restrict__ rowMax, float* __restrict__ rowSum) {
  int wave = threadIdx.x >> 5;
  int lane = threadIdx.x & 31;
  int n0 = (blockIdx.x * 4 + wave) * 16;
  int h  = blockIdx.y;
  int rh = lane >> 4;

  v16bf aq0 = load_frag_bf16(Qp, E_DIM, n0, h * D_DIM);
  v16bf aq1 = load_frag_bf16(Qp, E_DIM, n0, h * D_DIM + 32);

  float rmax[8], rsum[8];
#pragma unroll
  for (int v = 0; v < 8; ++v) { rmax[v] = -1.0e30f; rsum[v] = 0.0f; }

  v16bf bk00 = load_frag_bf16(Kp, E_DIM, 0, h * D_DIM);
  v16bf bk01 = load_frag_bf16(Kp, E_DIM, 0, h * D_DIM + 32);

  for (int s0 = 0; s0 < N_TOK; s0 += 32) {
    v16bf bk10 = load_frag_bf16(Kp, E_DIM, s0 + 16, h * D_DIM);
    v16bf bk11 = load_frag_bf16(Kp, E_DIM, s0 + 16, h * D_DIM + 32);
    v8f c = {};
    c = wmma_bf16(aq0, bk00, c);
    c = wmma_bf16(aq1, bk01, c);
    online_update(c, rmax, rsum);

    int sn = (s0 + 32) & (N_TOK - 1);      // last iter wraps (discarded)
    bk00 = load_frag_bf16(Kp, E_DIM, sn, h * D_DIM);
    bk01 = load_frag_bf16(Kp, E_DIM, sn, h * D_DIM + 32);
    v8f c2 = {};
    c2 = wmma_bf16(aq0, bk10, c2);
    c2 = wmma_bf16(aq1, bk11, c2);
    online_update(c2, rmax, rsum);
  }
  if ((lane & 15) == 0) {
    int base = h * N_TOK + n0 + 8 * rh;
#pragma unroll
    for (int v = 0; v < 8; ++v) { rowMax[base + v] = rmax[v]; rowSum[base + v] = rsum[v]; }
  }
}

// ---- pass 2: P tiles; O += P*V (WMMA); avg_w reduced in LDS, no atomics ---
// Block = (b, 16-row t-tile); its 16 waves are exactly the 16 (h, n-offset)
// combos contributing to avg_w[b, t0..t0+15, :]:
//   wave w -> h = 8b + (w&7), rows n0 = t0 + 1024*(w>>3).
// Each wave's private fp32 LDS slice holds its P tile (also the staging
// buffer for the P -> A-fragment relayout feeding the P*V WMMA).
__global__ __launch_bounds__(512) void attn_av_kernel(
    const bf16_t* __restrict__ Qp, const bf16_t* __restrict__ Kp,
    const bf16_t* __restrict__ Vt, const float* __restrict__ rowMax,
    const float* __restrict__ rowSum, bf16_t* __restrict__ Obf,
    float* __restrict__ avg_w) {
  __shared__ float slices[16][16 * 36];     // 16 waves x (16x32 P tile, pad 36)
  int wave = threadIdx.x >> 5;
  int lane = threadIdx.x & 31;
  int t0 = blockIdx.x * 16;
  int b  = blockIdx.y;
  int h  = 8 * b + (wave & 7);
  int n0 = t0 + 1024 * (wave >> 3);
  int col = lane & 15;
  int rh  = lane >> 4;
  float* myslice = slices[wave];

  v16bf aq0 = load_frag_bf16(Qp, E_DIM, n0, h * D_DIM);
  v16bf aq1 = load_frag_bf16(Qp, E_DIM, n0, h * D_DIM + 32);

  float mymax[8], myinv[8];
#pragma unroll
  for (int v = 0; v < 8; ++v) {
    int idx = h * N_TOK + n0 + v + 8 * rh;
    mymax[v] = rowMax[idx];
    myinv[v] = 1.0f / rowSum[idx];
  }

  const bf16_t* VtH = Vt + (size_t)h * D_DIM * N_TOK;
  v8f accO[4] = {};

  for (int s0 = 0; s0 < N_TOK; s0 += 32) {
    // Issue ALL global fragment loads for this s-chunk up front: 4 K-tile
    // frags (two 16-col score tiles) + 4 V frags; WMMA/VALU overlaps them.
    v16bf bk00 = load_frag_bf16(Kp, E_DIM, s0,      h * D_DIM);
    v16bf bk01 = load_frag_bf16(Kp, E_DIM, s0,      h * D_DIM + 32);
    v16bf bk10 = load_frag_bf16(Kp, E_DIM, s0 + 16, h * D_DIM);
    v16bf bk11 = load_frag_bf16(Kp, E_DIM, s0 + 16, h * D_DIM + 32);
    v16bf bv[4];
#pragma unroll
    for (int dt = 0; dt < 4; ++dt)
      bv[dt] = load_frag_bf16(VtH, N_TOK, dt * 16, s0);

#pragma unroll
    for (int t2 = 0; t2 < 2; ++t2) {
      v8f c = {};
      c = wmma_bf16(aq0, t2 ? bk10 : bk00, c);
      c = wmma_bf16(aq1, t2 ? bk11 : bk01, c);
#pragma unroll
      for (int v = 0; v < 8; ++v) {
        float p = __expf(c[v] - mymax[v]) * myinv[v];
        myslice[(v + 8 * rh) * 36 + t2 * 16 + col] = p;
      }
    }
    // Own-slice reads (same-wave DS ops are in-order): P as A-fragment.
    v16bf pa = load_frag_lds_f32(myslice, 36);
#pragma unroll
    for (int dt = 0; dt < 4; ++dt) accO[dt] = wmma_bf16(pa, bv[dt], accO[dt]);

    __syncthreads();                        // all 16 slices complete
    // avg_w flush: thread (wave,lane) owns element (row=wave, col=lane);
    // sum the 16 slices, one coalesced store, no atomics (tile is exclusive).
    float s = 0.0f;
#pragma unroll
    for (int w = 0; w < 16; ++w) s += slices[w][wave * 36 + lane];
    avg_w[((size_t)(b * T_DIM + t0 + wave)) * N_TOK + s0 + lane] = s * 0.0625f;
    __syncthreads();                        // before slices are overwritten
  }

#pragma unroll
  for (int dt = 0; dt < 4; ++dt) {
#pragma unroll
    for (int v = 0; v < 8; ++v) {
      int n = n0 + v + 8 * rh;
      int e = h * D_DIM + dt * 16 + col;
      Obf[(size_t)n * E_DIM + e] = (bf16_t)accO[dt][v];
    }
  }
}

// ------------------------------- out-proj ----------------------------------
// 16x64 per-wave register block; ping-pong double-buffered fragments.
__global__ __launch_bounds__(128) void out_proj_kernel(
    const bf16_t* __restrict__ Obf, const bf16_t* __restrict__ Wbf,
    const float* __restrict__ b_out, float* __restrict__ out) {
  int wave = threadIdx.x >> 5;
  int lane = threadIdx.x & 31;
  int e0 = (blockIdx.x * 4 + wave) * 64;
  int n0 = blockIdx.y * 16;

  v8f acc[4] = {};
  v16bf a0 = load_frag_bf16(Obf, E_DIM, n0, 0);
  v16bf b0[4];
#pragma unroll
  for (int et = 0; et < 4; ++et) b0[et] = load_frag_bf16(Wbf, E_DIM, e0 + et * 16, 0);

  for (int k0 = 0; k0 < E_DIM; k0 += 64) {
    v16bf a1 = load_frag_bf16(Obf, E_DIM, n0, k0 + 32);
    v16bf b1[4];
#pragma unroll
    for (int et = 0; et < 4; ++et)
      b1[et] = load_frag_bf16(Wbf, E_DIM, e0 + et * 16, k0 + 32);
#pragma unroll
    for (int et = 0; et < 4; ++et) acc[et] = wmma_bf16(a0, b0[et], acc[et]);

    int kn = (k0 + 64) & (E_DIM - 1);
    a0 = load_frag_bf16(Obf, E_DIM, n0, kn);
#pragma unroll
    for (int et = 0; et < 4; ++et)
      b0[et] = load_frag_bf16(Wbf, E_DIM, e0 + et * 16, kn);
#pragma unroll
    for (int et = 0; et < 4; ++et) acc[et] = wmma_bf16(a1, b1[et], acc[et]);
  }

  int col = lane & 15;
  int rh  = lane >> 4;
#pragma unroll
  for (int et = 0; et < 4; ++et) {
    int e = e0 + et * 16 + col;
    float bias = b_out[e];
#pragma unroll
    for (int v = 0; v < 8; ++v) {
      int n = n0 + v + 8 * rh;
      out[(size_t)n * E_DIM + e] = acc[et][v] + bias;
    }
  }
}

// ------------------------------- launcher ----------------------------------
extern "C" void kernel_launch(void* const* d_in, const int* in_sizes, int n_in,
                              void* d_out, int out_size, void* d_ws, size_t ws_size,
                              hipStream_t stream) {
  const float* query = (const float*)d_in[0];
  const float* key   = (const float*)d_in[1];
  // d_in[2] (value) intentionally unused: reference derives k AND v from `key`.
  const float* w_in  = (const float*)d_in[3];
  const float* b_in  = (const float*)d_in[4];
  const float* w_out = (const float*)d_in[5];
  const float* b_out = (const float*)d_in[6];

  float* out   = (float*)d_out;                                  // [T,B,E]
  float* avg_w = (float*)d_out + (size_t)T_DIM * B_DIM * E_DIM;  // [B,T,N_TOK]

  const size_t MB = 1024 * 1024;
  char* ws = (char*)d_ws;
  bf16_t* Qin   = (bf16_t*)(ws + 0 * MB);   // [N,E]   bf16 query
  bf16_t* Kin   = (bf16_t*)(ws + 4 * MB);   // [N,E]   bf16 key
  bf16_t* Winb  = (bf16_t*)(ws + 8 * MB);   // [3E,E]  bf16 in-proj weight
  bf16_t* Woutb = (bf16_t*)(ws + 14 * MB);  // [E,E]   bf16 out-proj weight
  bf16_t* Qp    = (bf16_t*)(ws + 16 * MB);  // [N,E]   projected q (scaled)
  bf16_t* Kp    = (bf16_t*)(ws + 20 * MB);  // [N,E]   projected k
  bf16_t* Vt    = (bf16_t*)(ws + 24 * MB);  // [H,D,N] projected v, transposed
  bf16_t* Obf   = (bf16_t*)(ws + 28 * MB);  // [N,E]   attention output
  float*  rMax  = (float*)(ws + 32 * MB);   // [H*N]
  float*  rSum  = (float*)(ws + 32 * MB + 256 * 1024);

  const int nQ  = N_TOK * E_DIM;          // 2M
  const int nW  = 3 * E_DIM * E_DIM;      // 3M
  const int nWo = E_DIM * E_DIM;          // 1M

  // 0) precision conversion (bf16 activations/weights; everything L2-resident)
  cvt_f32_to_bf16_kernel<<<nQ / 1024, 256, 0, stream>>>(query, Qin, nQ);
  cvt_f32_to_bf16_kernel<<<nQ / 1024, 256, 0, stream>>>(key, Kin, nQ);
  cvt_f32_to_bf16_kernel<<<nW / 1024, 256, 0, stream>>>(w_in, Winb, nW);
  cvt_f32_to_bf16_kernel<<<nWo / 1024, 256, 0, stream>>>(w_out, Woutb, nWo);

  // 1) fused QKV projection: [2048 x 3072] = X @ w_in^T (+bias, q scaled)
  proj_qkv_kernel<<<dim3(12, 128), 128, 0, stream>>>(Qin, Kin, Winb, b_in,
                                                     Qp, Kp, Vt);
  // 2) softmax row stats (online max/sumexp), one wave per (h, 16 rows)
  attn_stats_kernel<<<dim3(32, 16), 128, 0, stream>>>(Qp, Kp, rMax, rSum);
  // 3) recompute scores, normalize, O += P*V, avg_w LDS-reduced (no atomics)
  attn_av_kernel<<<dim3(64, 2), 512, 0, stream>>>(Qp, Kp, Vt, rMax, rSum,
                                                  Obf, avg_w);
  // 4) out-proj: out = O @ w_out^T + b_out
  out_proj_kernel<<<dim3(4, 128), 128, 0, stream>>>(Obf, Woutb, b_out, out);
}
